// FusionTokenRoutedMLP_10548439679096
// MI455X (gfx1250) — matrix operand order repaired and checked by reading, not compiled, not run on hardware
//
#include <hip/hip_runtime.h>
#include <math.h>

// ---- problem constants (match reference: B=8, N=2048, H=2048, E=4, I=2048) ----
#define HDIM   2048          // hidden size (K of GEMM1, N of GEMM2)
#define IDIM   2048          // intermediate per expert (N of GEMM1 half, K of GEMM2)
#define LDGU   4096          // gate_up_proj row stride (2*I)
#define NEXP   4
#define TOKPE  4096          // tokens per expert (B*N/E)
#define ROWPAD 20            // LDS row stride in dwords (16 data dwords + pad, mult of 4)

typedef __bf16 v16bf __attribute__((ext_vector_type(16)));
typedef float  v8f   __attribute__((ext_vector_type(8)));

union FragU { unsigned u[8]; v16bf v; };

__device__ __forceinline__ unsigned f32_to_bf16_rne(float f) {
    unsigned u = __float_as_uint(f);
    unsigned r = 0x7FFFu + ((u >> 16) & 1u);
    return (u + r) >> 16;
}

// split two f32 into packed bf16 hi (truncated) and lo (residual, RNE)
__device__ __forceinline__ void split2(float x0, float x1, unsigned &hi, unsigned &lo) {
    unsigned u0 = __float_as_uint(x0) & 0xFFFF0000u;
    unsigned u1 = __float_as_uint(x1) & 0xFFFF0000u;
    hi = (u0 >> 16) | u1;
    lo = f32_to_bf16_rne(x0 - __uint_as_float(u0)) |
         (f32_to_bf16_rne(x1 - __uint_as_float(u1)) << 16);
}

// A fragment: lane = c*16 + r holds row r; vgpr v holds K = (v&3)*2 + (v>>2)*16 + c*8 (+0,+1)
__device__ __forceinline__ v16bf load_frag_a(const unsigned* lds, int r, int c) {
    const uint4 x = *(const uint4*)(lds + r * ROWPAD + c * 4);
    const uint4 y = *(const uint4*)(lds + r * ROWPAD + 8 + c * 4);
    FragU t;
    t.u[0]=x.x; t.u[1]=x.y; t.u[2]=x.z; t.u[3]=x.w;
    t.u[4]=y.x; t.u[5]=y.y; t.u[6]=y.z; t.u[7]=y.w;
    return t.v;
}

// B fragment: lane = c*16 + n holds col n; vgpr v holds K = (v&3)*2 + (v>>2)*16 + c*8 (+0,+1)
__device__ __forceinline__ v16bf load_frag_b(const unsigned* lds, int n, int c) {
    const uint4 x = *(const uint4*)(lds + n * ROWPAD + c * 4);
    const uint4 y = *(const uint4*)(lds + n * ROWPAD + 8 + c * 4);
    FragU t;
    t.u[0]=x.x; t.u[1]=x.y; t.u[2]=x.z; t.u[3]=x.w;
    t.u[4]=y.x; t.u[5]=y.y; t.u[6]=y.z; t.u[7]=y.w;
    return t.v;
}

__device__ __forceinline__ v8f wmma_bf16(v16bf a, v16bf b, v8f c) {
    return __builtin_amdgcn_wmma_f32_16x16x32_bf16(false, a, false, b, (short)0, c, false, false);
}

// split-precision accumulate: acc += (aH+aL) x (bH+bL), dropping aL*bL
__device__ __forceinline__ v8f mm3(v16bf aH, v16bf aL, v16bf bH, v16bf bL, v8f acc) {
    acc = wmma_bf16(aH, bH, acc);
    acc = wmma_bf16(aH, bL, acc);
    acc = wmma_bf16(aL, bH, acc);
    return acc;
}

// =============== GEMM1: inter = silu(x@Wg) * (x@Wu), per expert ===============
// grid: (32 nBlk, 32 mBlk, 4 expert), block 256 threads = 8 waves (4 M x 2 N)
// block tile: M=128 expert-tokens, N=64 gate cols (+ matching 64 up cols)
__global__ __launch_bounds__(256) void moe_gemm1(
    const float* __restrict__ x, const float* __restrict__ gup,
    float* __restrict__ interm)
{
    __shared__ unsigned ldsA[2][128 * ROWPAD];   // [hi/lo], rows = tokens
    __shared__ unsigned ldsB[2][128 * ROWPAD];   // rows 0..63 gate cols, 64..127 up cols

    const int e    = blockIdx.z;
    const int mBlk = blockIdx.y;
    const int nBlk = blockIdx.x;
    const int tid  = threadIdx.x;
    const int lane = tid & 31;
    const int wave = tid >> 5;
    const int wm   = wave >> 1;          // 0..3
    const int wn   = wave & 1;           // 0..1
    const int c    = lane >> 4;          // K-half select
    const int rl   = lane & 15;          // row/col within 16

    const float* W = gup + (size_t)e * HDIM * LDGU;
    const int mA  = tid >> 4;            // A staging: 16 rows per pass
    const int kpA = tid & 15;            //           16 k-pairs
    const int nB  = tid & 63;            // B staging: 64 cols
    const int kqB = tid >> 6;            //           4 k-pairs per pass
    const size_t cg = (size_t)nBlk * 64 + nB;

    // k-invariant gathered token row offsets for A staging
    size_t rowOff[8];
    #pragma unroll
    for (int p = 0; p < 8; ++p) {
        int mg = mBlk * 128 + mA + p * 16;
        int b  = mg >> 9, g = mg & 511;
        rowOff[p] = ((size_t)b * 2048 + (size_t)g * 4 + e) * HDIM + kpA * 2;
    }

    v8f accG[2][2], accU[2][2];
    #pragma unroll
    for (int i = 0; i < 2; ++i)
        #pragma unroll
        for (int j = 0; j < 2; ++j) { accG[i][j] = {}; accU[i][j] = {}; }

    // ---- prologue: prefetch tile k0=0 into registers ----
    float2 aP[8];
    float bg0[4], bg1[4], bu0[4], bu1[4];
    #pragma unroll
    for (int p = 0; p < 8; ++p) aP[p] = *(const float2*)(x + rowOff[p]);
    #pragma unroll
    for (int p = 0; p < 4; ++p) {
        size_t k = (size_t)(kqB + p * 4) * 2;
        bg0[p] = W[k * LDGU + cg];        bg1[p] = W[(k + 1) * LDGU + cg];
        bu0[p] = W[k * LDGU + cg + IDIM]; bu1[p] = W[(k + 1) * LDGU + cg + IDIM];
    }

    for (int k0 = 0; k0 < HDIM; k0 += 32) {
        __syncthreads();                 // previous tile's LDS readers done
        // ---- regs -> LDS (split to bf16 hi/lo) ----
        #pragma unroll
        for (int p = 0; p < 8; ++p) {
            unsigned hi, lo; split2(aP[p].x, aP[p].y, hi, lo);
            ldsA[0][(mA + p * 16) * ROWPAD + kpA] = hi;
            ldsA[1][(mA + p * 16) * ROWPAD + kpA] = lo;
        }
        #pragma unroll
        for (int p = 0; p < 4; ++p) {
            int kq = kqB + p * 4;
            unsigned hi, lo;
            split2(bg0[p], bg1[p], hi, lo);
            ldsB[0][nB * ROWPAD + kq] = hi;
            ldsB[1][nB * ROWPAD + kq] = lo;
            split2(bu0[p], bu1[p], hi, lo);
            ldsB[0][(64 + nB) * ROWPAD + kq] = hi;
            ldsB[1][(64 + nB) * ROWPAD + kq] = lo;
        }
        __syncthreads();
        // ---- prefetch next tile (overlaps with WMMA below) ----
        if (k0 + 32 < HDIM) {
            int kn = k0 + 32;
            #pragma unroll
            for (int p = 0; p < 8; ++p) aP[p] = *(const float2*)(x + rowOff[p] + kn);
            #pragma unroll
            for (int p = 0; p < 4; ++p) {
                size_t k = (size_t)kn + (size_t)(kqB + p * 4) * 2;
                bg0[p] = W[k * LDGU + cg];        bg1[p] = W[(k + 1) * LDGU + cg];
                bu0[p] = W[k * LDGU + cg + IDIM]; bu1[p] = W[(k + 1) * LDGU + cg + IDIM];
            }
        }
        // ---- compute ----
        v16bf aH[2], aL[2];
        #pragma unroll
        for (int ms = 0; ms < 2; ++ms) {
            int r = wm * 32 + ms * 16 + rl;
            aH[ms] = load_frag_a(ldsA[0], r, c);
            aL[ms] = load_frag_a(ldsA[1], r, c);
        }
        #pragma unroll
        for (int ns = 0; ns < 2; ++ns) {
            int n = wn * 32 + ns * 16 + rl;
            v16bf bH = load_frag_b(ldsB[0], n, c);
            v16bf bL = load_frag_b(ldsB[1], n, c);
            #pragma unroll
            for (int ms = 0; ms < 2; ++ms)
                accG[ms][ns] = mm3(aH[ms], aL[ms], bH, bL, accG[ms][ns]);
            v16bf uH = load_frag_b(ldsB[0], 64 + n, c);
            v16bf uL = load_frag_b(ldsB[1], 64 + n, c);
            #pragma unroll
            for (int ms = 0; ms < 2; ++ms)
                accU[ms][ns] = mm3(aH[ms], aL[ms], uH, uL, accU[ms][ns]);
        }
    }

    // ---- epilogue: inter = silu(gate)*up, f32 to workspace ----
    const int mBase = mBlk * 128 + wm * 32;
    const size_t rowBase = (size_t)e * TOKPE;
    #pragma unroll
    for (int ms = 0; ms < 2; ++ms)
        #pragma unroll
        for (int ns = 0; ns < 2; ++ns) {
            int col = nBlk * 64 + wn * 32 + ns * 16 + rl;
            #pragma unroll
            for (int v = 0; v < 8; ++v) {
                int m = mBase + ms * 16 + v + 8 * c;
                float g = accG[ms][ns][v];
                float u = accU[ms][ns][v];
                float s = (g / (1.0f + __expf(-g))) * u;
                interm[(rowBase + m) * IDIM + col] = s;
            }
        }
}

// =============== GEMM2: out = inter @ down_proj[e], scatter to tokens ===============
// grid: (16 nBlk, 32 mBlk, 4 expert); block tile M=128, N=128; wave tile 32x64
__global__ __launch_bounds__(256) void moe_gemm2(
    const float* __restrict__ interm, const float* __restrict__ down,
    float* __restrict__ out)
{
    __shared__ unsigned ldsA[2][128 * ROWPAD];
    __shared__ unsigned ldsB[2][128 * ROWPAD];

    const int e    = blockIdx.z;
    const int mBlk = blockIdx.y;
    const int nBlk = blockIdx.x;
    const int tid  = threadIdx.x;
    const int lane = tid & 31;
    const int wave = tid >> 5;
    const int wm   = wave >> 1;          // 0..3
    const int wn   = wave & 1;           // 0..1
    const int c    = lane >> 4;
    const int rl   = lane & 15;

    const float* W = down + (size_t)e * IDIM * HDIM;
    const float* A = interm + (size_t)e * TOKPE * IDIM;
    const int mA = tid >> 4, kpA = tid & 15;
    const int nB = tid & 127;            // B staging: 128 cols
    const int kqB = tid >> 7;            //           2 k-pairs per pass
    const size_t cg = (size_t)nBlk * 128 + nB;

    v8f acc[2][4];
    #pragma unroll
    for (int i = 0; i < 2; ++i)
        #pragma unroll
        for (int j = 0; j < 4; ++j) acc[i][j] = {};

    // ---- prologue prefetch ----
    float2 aP[8];
    float b0[8], b1[8];
    #pragma unroll
    for (int p = 0; p < 8; ++p)
        aP[p] = *(const float2*)(A + (size_t)(mBlk * 128 + mA + p * 16) * IDIM + kpA * 2);
    #pragma unroll
    for (int p = 0; p < 8; ++p) {
        size_t k = (size_t)(kqB + p * 2) * 2;
        b0[p] = W[k * HDIM + cg];
        b1[p] = W[(k + 1) * HDIM + cg];
    }

    for (int k0 = 0; k0 < IDIM; k0 += 32) {
        __syncthreads();
        #pragma unroll
        for (int p = 0; p < 8; ++p) {
            unsigned hi, lo; split2(aP[p].x, aP[p].y, hi, lo);
            ldsA[0][(mA + p * 16) * ROWPAD + kpA] = hi;
            ldsA[1][(mA + p * 16) * ROWPAD + kpA] = lo;
        }
        #pragma unroll
        for (int p = 0; p < 8; ++p) {
            int kq = kqB + p * 2;
            unsigned hi, lo; split2(b0[p], b1[p], hi, lo);
            ldsB[0][nB * ROWPAD + kq] = hi;
            ldsB[1][nB * ROWPAD + kq] = lo;
        }
        __syncthreads();
        if (k0 + 32 < IDIM) {
            int kn = k0 + 32;
            #pragma unroll
            for (int p = 0; p < 8; ++p)
                aP[p] = *(const float2*)(A + (size_t)(mBlk * 128 + mA + p * 16) * IDIM + kn + kpA * 2);
            #pragma unroll
            for (int p = 0; p < 8; ++p) {
                size_t k = (size_t)kn + (size_t)(kqB + p * 2) * 2;
                b0[p] = W[k * HDIM + cg];
                b1[p] = W[(k + 1) * HDIM + cg];
            }
        }
        v16bf aH[2], aL[2];
        #pragma unroll
        for (int ms = 0; ms < 2; ++ms) {
            int r = wm * 32 + ms * 16 + rl;
            aH[ms] = load_frag_a(ldsA[0], r, c);
            aL[ms] = load_frag_a(ldsA[1], r, c);
        }
        #pragma unroll
        for (int ns = 0; ns < 4; ++ns) {
            int n = wn * 64 + ns * 16 + rl;
            v16bf bH = load_frag_b(ldsB[0], n, c);
            v16bf bL = load_frag_b(ldsB[1], n, c);
            #pragma unroll
            for (int ms = 0; ms < 2; ++ms)
                acc[ms][ns] = mm3(aH[ms], aL[ms], bH, bL, acc[ms][ns]);
        }
    }

    // scatter rows back to token order: t = b*2048 + g*4 + e
    #pragma unroll
    for (int ms = 0; ms < 2; ++ms)
        #pragma unroll
        for (int ns = 0; ns < 4; ++ns) {
            int col = nBlk * 128 + wn * 64 + ns * 16 + rl;
            #pragma unroll
            for (int v = 0; v < 8; ++v) {
                int mg = mBlk * 128 + wm * 32 + ms * 16 + v + 8 * c;
                int b = mg >> 9, g = mg & 511;
                size_t t = (size_t)b * 2048 + (size_t)g * 4 + e;
                out[t * HDIM + col] = acc[ms][ns][v];
            }
        }
}

extern "C" void kernel_launch(void* const* d_in, const int* in_sizes, int n_in,
                              void* d_out, int out_size, void* d_ws, size_t ws_size,
                              hipStream_t stream) {
    const float* x    = (const float*)d_in[0];
    const float* gup  = (const float*)d_in[1];
    const float* down = (const float*)d_in[2];
    float* out    = (float*)d_out;
    float* interm = (float*)d_ws;   // 16384 * 2048 f32 = 128 MB

    dim3 block(256);
    dim3 grid1(IDIM / 64,  TOKPE / 128, NEXP);   // 32 x 32 x 4
    dim3 grid2(HDIM / 128, TOKPE / 128, NEXP);   // 16 x 32 x 4
    moe_gemm1<<<grid1, block, 0, stream>>>(x, gup, interm);
    moe_gemm2<<<grid2, block, 0, stream>>>(interm, down, out);
}
